// SuperResolutionModel_69252052681011
// MI455X (gfx1250) — matrix operand order repaired and checked by reading, not compiled, not run on hardware
//
#include <hip/hip_runtime.h>

typedef __attribute__((ext_vector_type(16))) _Float16 v16h;
typedef __attribute__((ext_vector_type(8)))  _Float16 v8h;
typedef __attribute__((ext_vector_type(8)))  float    v8f;

#define IN_H 39
#define IN_W 88
#define C_IN 17
#define NF   10
#define OUT_H 156
#define OUT_W 352
#define NB   64
#define D1   154   // 3*3*17+1
#define D2   361   // 6*6*10+1

// workspace layout (bytes, all >=32B aligned)
static constexpr size_t OFF_W1  = 0;        // w1 padded [153][16] f32
static constexpr size_t OFF_W2T = 16384;    // w2 [360][10] f32 (p=(dy*6+dx)*10+c, f)
static constexpr size_t OFF_W3  = 32768;    // w3 padded [4][16] f32
static constexpr size_t OFF_ZB  = 33280;    // 64 B zeros (A-matrix OOB source)
static constexpr size_t OFF_BPK = 36864;    // B matrices [16 phase][2 wmma][32 lane][16 h] f16
static constexpr size_t OFF_H1  = 69632;    // h1 f16 [NB][39][88][16]
static constexpr size_t OFF_H2  = 7098368;  // h2 f16 [NB][156][352][16]

// Upsample phase tables: for phase p, first/second tap kernel offsets dy (or dx).
__device__ __forceinline__ void dtaps(int p, int& d0, int& d1) {
    d0 = (p == 0) ? 0 : (p == 1) ? 3 : (p == 2) ? 2 : 1;
    d1 = (p == 0) ? 4 : (p == 3) ? 5 : -1;       // -1 = no second tap
}
__device__ __forceinline__ void otaps(int p, int& o0, int& o1) {
    o0 = (p == 0) ? -1 : 0;                      // h1 index offset from q
    o1 = (p == 0) ? 0 : ((p == 3) ? 1 : 999);    // 999 = invalid
}

// ---------------- Stage 0: horseshoe weights via rank-1 Cholesky recurrence ----
__global__ void weights_kernel(
    const float* __restrict__ Up1, const float* __restrict__ Uh1,
    const float* __restrict__ ta1, const float* __restrict__ ep1,
    const float* __restrict__ nu1,
    const float* __restrict__ Up2, const float* __restrict__ Uh2,
    const float* __restrict__ ta2, const float* __restrict__ ep2,
    const float* __restrict__ nu2,
    const float* __restrict__ kap3, const float* __restrict__ ep3,
    const float* __restrict__ cp,
    float* __restrict__ w1p, float* __restrict__ w2T,
    float* __restrict__ w3p, float* __restrict__ zb,
    _Float16* __restrict__ bpk)
{
    const int tid = threadIdx.x;  // 64 threads
    const float cc = cp[0];

    if (tid < NF) {
        // filter tid of layer 1: L*eps for U = diag(d) + v v^T in O(D1)
        const int f = tid;
        const float* d = Up1 + f * D1;
        const float* v = Uh1 + f * D1;
        const float* e = ep1 + f * D1;
        float u = 1.f, P = 0.f, sl = 0.f;
        for (int k = 0; k < D1; ++k) {
            float dk = d[k], vk = v[k], ek = e[k];
            float diag = dk + vk * vk * u;
            float Lkk  = sqrtf(diag);
            float s    = vk * P + Lkk * ek;
            if (k < D1 - 1) w1p[k * 16 + f] = s; else sl = s;
            float beta = u / Lkk;
            P += vk * beta * ek;
            u  = u * dk / diag;
        }
        float tau = expf(sl + logf(ta1[f]));
        float nu  = nu1[0];
        float t2 = tau * tau, c2 = cc * cc;
        float scale = sqrtf(c2 * t2 / (c2 + t2 * nu * nu)) * nu;
        for (int k = 0; k < D1 - 1; ++k) w1p[k * 16 + f] *= scale;
    } else if (tid >= 10 && tid < 16) {
        for (int k = 0; k < D1 - 1; ++k) w1p[k * 16 + tid] = 0.f;  // pad filters
    } else if (tid >= 16 && tid < 16 + NF) {
        const int f = tid - 16;
        const float* d = Up2 + f * D2;
        const float* v = Uh2 + f * D2;
        const float* e = ep2 + f * D2;
        float u = 1.f, P = 0.f, sl = 0.f;
        for (int k = 0; k < D2; ++k) {
            float dk = d[k], vk = v[k], ek = e[k];
            float diag = dk + vk * vk * u;
            float Lkk  = sqrtf(diag);
            float s    = vk * P + Lkk * ek;
            if (k < D2 - 1) w2T[k * NF + f] = s; else sl = s;
            float beta = u / Lkk;
            P += vk * beta * ek;
            u  = u * dk / diag;
        }
        float tau = expf(sl + logf(ta2[f]));
        float nu  = nu2[0];
        float t2 = tau * tau, c2 = cc * cc;
        float scale = sqrtf(c2 * t2 / (c2 + t2 * nu * nu)) * nu;
        for (int k = 0; k < D2 - 1; ++k) w2T[k * NF + f] *= scale;
    } else if (tid == 32) {
        for (int i = 0; i < 64; ++i) {          // w3 padded [tap(4)][16]
            int tap = i >> 4, c = i & 15;
            float v = 0.f;
            if (c < NF) { int p = tap * NF + c; float ka = kap3[p]; v = ka * ka * ep3[p]; }
            w3p[i] = v;
        }
    } else if (tid == 33) {
        for (int i = 0; i < 16; ++i) zb[i] = 0.f;  // zero A-source
    }
    __syncthreads();

    // Pack per-phase WMMA B matrices in wave32 B lane layout:
    // bpk[ph][w][lane][h], N = lane%16, K = (h&7) + 8*(lane>=16) + 16*(h>=8) + 32*w,
    // K = tap*16 + c, tap = ty*2+tx.
    for (int idx = tid; idx < 16384; idx += 64) {
        int ph = idx >> 10, rem = idx & 1023;
        int w = rem >> 9, lane = (rem >> 4) & 31, h = rem & 15;
        int n = lane & 15;
        int k = (h & 7) + ((lane >> 4) << 3) + ((h >> 3) << 4) + (w << 5);
        int tap = k >> 4, c = k & 15;
        int ty = tap >> 1, tx = tap & 1;
        int py = ph >> 2, px = ph & 3;
        int dy0, dy1, dx0, dx1;
        dtaps(py, dy0, dy1);
        dtaps(px, dx0, dx1);
        int dy = ty ? dy1 : dy0;
        int dx = tx ? dx1 : dx0;
        float val = 0.f;
        if (dy >= 0 && dx >= 0 && c < NF && n < NF)
            val = w2T[((dy * 6 + dx) * NF + c) * NF + n];
        bpk[idx] = (_Float16)val;
    }
}

// ---------------- Stage 1: conv1 3x3x17->10 + relu, write f16 channel-padded ---
__global__ void __launch_bounds__(256) conv1_kernel(
    const float* __restrict__ x, const float* __restrict__ w1p,
    _Float16* __restrict__ h1)
{
    int gid = blockIdx.x * 256 + threadIdx.x;        // exact: NB*39*88
    int j = gid % IN_W;
    int t = gid / IN_W;
    int i = t % IN_H;
    int b = t / IN_H;
    float acc[NF];
#pragma unroll
    for (int f = 0; f < NF; ++f) acc[f] = 0.f;
    for (int dy = 0; dy < 3; ++dy) {
        int ii = i + dy - 1;
        if ((unsigned)ii >= (unsigned)IN_H) continue;
        for (int dx = 0; dx < 3; ++dx) {
            int jj = j + dx - 1;
            if ((unsigned)jj >= (unsigned)IN_W) continue;
            const float* xp = x + (((size_t)b * IN_H + ii) * IN_W + jj) * C_IN;
            int kbase = (dy * 3 + dx) * C_IN;
            for (int ci = 0; ci < C_IN; ++ci) {
                float xv = xp[ci];
                const float* wr = w1p + (kbase + ci) * 16;   // uniform -> s_load
#pragma unroll
                for (int f = 0; f < NF; ++f) acc[f] += xv * wr[f];
            }
        }
    }
    _Float16 o[16];
#pragma unroll
    for (int f = 0; f < 16; ++f)
        o[f] = (_Float16)((f < NF) ? fmaxf(acc[f], 0.f) : 0.f);
    v8h* dst = (v8h*)(h1 + (size_t)gid * 16);
    dst[0] = *(const v8h*)&o[0];
    dst[1] = *(const v8h*)&o[8];
}

// ---------------- Stage 2: conv2 as phase-GEMM, 2x v_wmma_f32_16x16x32_f16 -----
// Block = 128 thr = 4 waves = 4 x-phases of one output row; wave does 16 pixels.
__global__ void __launch_bounds__(128) conv2_kernel(
    const _Float16* __restrict__ h1, const _Float16* __restrict__ bpk,
    const _Float16* __restrict__ zb, _Float16* __restrict__ h2)
{
    __shared__ _Float16 tile[64 * 16];
    const int tid  = threadIdx.x;
    const int px   = tid >> 5;           // wave index = x phase
    const int lane = tid & 31;
    const int t    = lane & 15;          // M (pixel) for A; N (filter) for D
    const int hi   = lane >> 4;
    const int xt = blockIdx.x, y = blockIdx.y, b = blockIdx.z;
    const int py = y & 3, qy = y >> 2;
    const int q  = xt * 16 + t;          // x = px + 4*q
    int io0, io1, jo0, jo1;
    otaps(py, io0, io1);
    otaps(px, jo0, jo1);
    const int i0 = qy + io0, i1 = qy + io1;
    const int j0 = q + jo0,  j1 = q + jo1;
    const bool qv = q < IN_W;

    auto ld = [&](int i, int j) -> v8h {
        bool ok = qv & ((unsigned)i < (unsigned)IN_H) & ((unsigned)j < (unsigned)IN_W);
        const _Float16* p = ok ? (h1 + (((size_t)b * IN_H + i) * IN_W + j) * 16 + hi * 8)
                               : zb;
        return *(const v8h*)p;
    };
    v8h L00 = ld(i0, j0), L01 = ld(i0, j1), L10 = ld(i1, j0), L11 = ld(i1, j1);
    union AU { v16h v; v8h h[2]; };
    AU a1, a2;
    a1.h[0] = L00; a1.h[1] = L01;   // K 0..31 : taps (y0,x0),(y0,x1)
    a2.h[0] = L10; a2.h[1] = L11;   // K32..63 : taps (y1,x0),(y1,x1)

    const int ph = py * 4 + px;
    const v16h* bp = (const v16h*)(bpk + (size_t)ph * 2 * 32 * 16);
    v16h B1 = bp[lane];
    v16h B2 = bp[32 + lane];

    v8f acc = {};
    acc = __builtin_amdgcn_wmma_f32_16x16x32_f16(false, a1.v, false, B1,
                                                 (short)0, acc, false, false);
    acc = __builtin_amdgcn_wmma_f32_16x16x32_f16(false, a2.v, false, B2,
                                                 (short)0, acc, false, false);

    // D: VGPR r -> M = r + 8*hi (pixel), lane%16 -> N (filter). ReLU except the
    // upsample-edge band (fixed up + activated by fixup_kernel).
#pragma unroll
    for (int r = 0; r < 8; ++r) {
        int M  = r + 8 * hi;
        int xl = px + 4 * M;
        int xg = xt * 64 + xl;
        float v = acc[r];
        v = ((y >= 152) | (xg >= 348)) ? v : fmaxf(v, 0.f);
        tile[xl * 16 + t] = (_Float16)v;
    }
    __syncthreads();

    // Coalesced writeback: 128 threads x 16B contiguous.
    int xl = tid >> 1, hf = (tid & 1) * 8;
    int xg = xt * 64 + xl;
    if (xg < OUT_W)
        *(v8h*)(h2 + (((size_t)b * OUT_H + y) * OUT_W + xg) * 16 + hf) =
            *(const v8h*)&tile[xl * 16 + hf];
}

// ---------------- Stage 2b: edge fixup (duplicated last upsample row/col) ------
__global__ void __launch_bounds__(256) fixup_kernel(
    const _Float16* __restrict__ h1, const float* __restrict__ w2T,
    _Float16* __restrict__ h2)
{
    int gid = blockIdx.x * 256 + threadIdx.x;
    if (gid >= NB * 2016) return;
    int b = gid / 2016, pos = gid % 2016;
    int y, x;
    if (pos < 1408) { y = 152 + pos / 352; x = pos % 352; }
    else            { int p2 = pos - 1408; y = p2 >> 2; x = 348 + (p2 & 3); }

    float corr[NF];
#pragma unroll
    for (int f = 0; f < NF; ++f) corr[f] = 0.f;

    auto add = [&](int i, int j, int dy, int dx) {
        const _Float16* hp = h1 + (((size_t)b * IN_H + i) * IN_W + j) * 16;
        const float* wb = w2T + (size_t)(dy * 6 + dx) * NF * NF;
        for (int c = 0; c < NF; ++c) {
            float v = (float)hp[c];
#pragma unroll
            for (int f = 0; f < NF; ++f) corr[f] += v * wb[c * NF + f];
        }
    };

    const bool ey = (y >= 152), ex = (x >= 348);
    const int py = y & 3, qy = y >> 2, px = x & 3, q = x >> 2;
    if (ey) {
        int dy = 157 - y;                            // duplicated row 155 -> h1 row 38
        int jo0, jo1, dx0, dx1;
        otaps(px, jo0, jo1);
        dtaps(px, dx0, dx1);
        int jj0 = q + jo0;
        if ((unsigned)jj0 < (unsigned)IN_W) add(38, jj0, dy, dx0);
        int jj1 = q + jo1;
        if (jo1 != 999 && (unsigned)jj1 < (unsigned)IN_W && dx1 >= 0) add(38, jj1, dy, dx1);
        if (ex) add(38, 87, dy, 353 - x);            // corner cross term
    }
    if (ex) {
        int dx = 353 - x;                            // duplicated col 351 -> h1 col 87
        int io0, io1, dy0, dy1;
        otaps(py, io0, io1);
        dtaps(py, dy0, dy1);
        int ii0 = qy + io0;
        if ((unsigned)ii0 < (unsigned)IN_H) add(ii0, 87, dy0, dx);
        int ii1 = qy + io1;
        if (io1 != 999 && (unsigned)ii1 < (unsigned)IN_H && dy1 >= 0) add(ii1, 87, dy1, dx);
    }

    _Float16* hp2 = h2 + (((size_t)b * OUT_H + y) * OUT_W + x) * 16;
#pragma unroll
    for (int f = 0; f < NF; ++f) {
        float v = (float)hp2[f] + corr[f];
        hp2[f] = (_Float16)fmaxf(v, 0.f);
    }
}

// ---------------- Stage 3: conv3 2x2x10->1 + relu -----------------------------
__global__ void __launch_bounds__(256) conv3_kernel(
    const _Float16* __restrict__ h2, const float* __restrict__ w3p,
    float* __restrict__ out)
{
    int gid = blockIdx.x * 256 + threadIdx.x;        // exact: NB*156*352
    int x = gid % OUT_W;
    int t = gid / OUT_W;
    int y = t % OUT_H;
    int b = t / OUT_H;
    float acc = 0.f;
#pragma unroll
    for (int dy = 0; dy < 2; ++dy) {
        int yy = y + dy;
        if (yy >= OUT_H) continue;
#pragma unroll
        for (int dx = 0; dx < 2; ++dx) {
            int xx = x + dx;
            if (xx >= OUT_W) continue;
            const v8h* hp = (const v8h*)(h2 + (((size_t)b * OUT_H + yy) * OUT_W + xx) * 16);
            v8h a0 = hp[0], a1 = hp[1];
            const float* wr = w3p + (dy * 2 + dx) * 16;  // uniform -> s_load
#pragma unroll
            for (int c = 0; c < 8; ++c) acc += (float)a0[c] * wr[c];
#pragma unroll
            for (int c = 0; c < 8; ++c) acc += (float)a1[c] * wr[8 + c];
        }
    }
    out[gid] = fmaxf(acc, 0.f);
}

extern "C" void kernel_launch(void* const* d_in, const int* in_sizes, int n_in,
                              void* d_out, int out_size, void* d_ws, size_t ws_size,
                              hipStream_t stream)
{
    (void)in_sizes; (void)n_in; (void)out_size; (void)ws_size;
    const float* x    = (const float*)d_in[0];
    const float* Up1  = (const float*)d_in[1];
    const float* Uh1  = (const float*)d_in[2];
    const float* ta1  = (const float*)d_in[3];
    const float* ep1  = (const float*)d_in[4];
    const float* nu1  = (const float*)d_in[5];
    const float* Up2  = (const float*)d_in[6];
    const float* Uh2  = (const float*)d_in[7];
    const float* ta2  = (const float*)d_in[8];
    const float* ep2  = (const float*)d_in[9];
    const float* nu2  = (const float*)d_in[10];
    const float* kap3 = (const float*)d_in[11];
    const float* ep3  = (const float*)d_in[12];
    const float* cc   = (const float*)d_in[13];

    char* ws = (char*)d_ws;
    float*    w1p = (float*)(ws + OFF_W1);
    float*    w2T = (float*)(ws + OFF_W2T);
    float*    w3p = (float*)(ws + OFF_W3);
    float*    zb  = (float*)(ws + OFF_ZB);
    _Float16* bpk = (_Float16*)(ws + OFF_BPK);
    _Float16* h1  = (_Float16*)(ws + OFF_H1);
    _Float16* h2  = (_Float16*)(ws + OFF_H2);

    weights_kernel<<<1, 64, 0, stream>>>(Up1, Uh1, ta1, ep1, nu1,
                                         Up2, Uh2, ta2, ep2, nu2,
                                         kap3, ep3, cc,
                                         w1p, w2T, w3p, zb, bpk);
    conv1_kernel<<<(NB * IN_H * IN_W) / 256, 256, 0, stream>>>(x, w1p, h1);
    conv2_kernel<<<dim3(6, OUT_H, NB), 128, 0, stream>>>(h1, bpk,
                                                         (const _Float16*)zb, h2);
    fixup_kernel<<<(NB * 2016 + 255) / 256, 256, 0, stream>>>(h1, w2T, h2);
    conv3_kernel<<<(NB * OUT_H * OUT_W) / 256, 256, 0, stream>>>(h2, w3p,
                                                                 (float*)d_out);
}
